// Attention_86758339379678
// MI455X (gfx1250) — matrix operand order repaired and checked by reading, not compiled, not run on hardware
//
#include <hip/hip_runtime.h>
#include <hip/hip_bf16.h>

// Flash-attention forward for B=4,H=16,S=2048,D=64 fp32, causal.
// f16 WMMA (v_wmma_f32_16x16x32_f16) with fp32 accumulation, online softmax.
// Row-sum of exp(P) computed by WMMA against an all-ones B fragment.

typedef __attribute__((ext_vector_type(16))) _Float16 v16h;
typedef __attribute__((ext_vector_type(8)))  _Float16 v8h;
typedef __attribute__((ext_vector_type(8)))  float    v8f;

#define S_LEN   2048
#define D_DIM   64
#define NBH     64      // B*H
#define BLOCK_M 128     // query rows per workgroup (8 waves x 16 rows)
#define KBLK    32      // key rows per iteration (two 16-wide WMMA N tiles)
#define VT_STRIDE 40    // halfs; 80B rows -> 16B-aligned b128 lds loads
#define PS_STRIDE 40

__device__ __forceinline__ float rmax16(float v) {
    v = fmaxf(v, __shfl_xor(v, 1, 32));
    v = fmaxf(v, __shfl_xor(v, 2, 32));
    v = fmaxf(v, __shfl_xor(v, 4, 32));
    v = fmaxf(v, __shfl_xor(v, 8, 32));
    return v;
}

__device__ __forceinline__ v16h pack16s(float4 f0, float4 f1, float4 f2, float4 f3,
                                        float s) {
    v16h a;
    a[0]  = (_Float16)(f0.x * s); a[1]  = (_Float16)(f0.y * s);
    a[2]  = (_Float16)(f0.z * s); a[3]  = (_Float16)(f0.w * s);
    a[4]  = (_Float16)(f1.x * s); a[5]  = (_Float16)(f1.y * s);
    a[6]  = (_Float16)(f1.z * s); a[7]  = (_Float16)(f1.w * s);
    a[8]  = (_Float16)(f2.x * s); a[9]  = (_Float16)(f2.y * s);
    a[10] = (_Float16)(f2.z * s); a[11] = (_Float16)(f2.w * s);
    a[12] = (_Float16)(f3.x * s); a[13] = (_Float16)(f3.y * s);
    a[14] = (_Float16)(f3.z * s); a[15] = (_Float16)(f3.w * s);
    return a;
}

__device__ __forceinline__ v8h pack8(float4 f0, float4 f1) {
    v8h h;
    h[0] = (_Float16)f0.x; h[1] = (_Float16)f0.y; h[2] = (_Float16)f0.z; h[3] = (_Float16)f0.w;
    h[4] = (_Float16)f1.x; h[5] = (_Float16)f1.y; h[6] = (_Float16)f1.z; h[7] = (_Float16)f1.w;
    return h;
}

__device__ __forceinline__ v8f wmma_f16(v16h a, v16h b, v8f c) {
    return __builtin_amdgcn_wmma_f32_16x16x32_f16(false, a, false, b, (short)0, c, false, false);
}

__global__ __launch_bounds__(256) void fa_fwd_kernel(
        const float* __restrict__ Q, const float* __restrict__ K,
        const float* __restrict__ V, float* __restrict__ O) {
    // K tile: row-major [32 keys][64 d] f16 (4 KB)
    __shared__ _Float16 kT[KBLK * D_DIM];
    // V tile: transposed [64 d][VT_STRIDE keys] f16 (5 KB)
    __shared__ _Float16 vT[D_DIM * VT_STRIDE];
    // per-wave P scratch: [16 rows][PS_STRIDE] f16
    __shared__ _Float16 pS[8][16 * PS_STRIDE];

    const int bh    = blockIdx.y;            // fused (b,h)
    const int qblk  = blockIdx.x;
    const int qbase = qblk * BLOCK_M;
    const int tid   = threadIdx.x;
    const int w     = tid >> 5;              // wave 0..7
    const int lane  = tid & 31;
    const int hf    = lane >> 4;             // half-wave
    const int ln    = lane & 15;
    const int qw    = qbase + w * 16;        // this wave's first query row

    const size_t base = (size_t)bh * S_LEN * D_DIM;
    const float* Qb = Q + base;
    const float* Kb = K + base;
    const float* Vb = V + base;
    float*       Ob = O + base;

    // ---- Q A-fragments, pre-scaled by 1/sqrt(D), held for the whole pass ----
    // A 16x32 k order chosen here (not the ISA table order): lane ln = row M; lanes<16 hold K 0-15,
    // lanes>=16 hold K 16-31 (contiguous, matching B fragments).
    v16h qa[2];
    {
        const int qrow = qw + ln;
#pragma unroll
        for (int s = 0; s < 2; ++s) {
            const float* qp = Qb + (size_t)qrow * D_DIM + s * 32 + hf * 16;
            float4 f0 = *(const float4*)(qp + 0);
            float4 f1 = *(const float4*)(qp + 4);
            float4 f2 = *(const float4*)(qp + 8);
            float4 f3 = *(const float4*)(qp + 12);
            qa[s] = pack16s(f0, f1, f2, f3, 0.125f);   // 1/sqrt(64)
        }
    }

    // all-ones B fragment: row-sum reducer for P (sum_k P[m][k]*1)
    v16h ones;
#pragma unroll
    for (int i = 0; i < 16; ++i) ones[i] = (_Float16)1.0f;

    float m[8], l[8];
    v8f   o[4] = {};    // 16 rows x 64 cols output accumulator
#pragma unroll
    for (int r = 0; r < 8; ++r) { m[r] = -1e30f; l[r] = 0.0f; }

    const int kbMax = qblk * 4 + 4;          // causal bound for this block
    const int krow  = tid >> 3;              // coop-load row (0..31)
    const int kcol  = (tid & 7) * 8;         // coop-load d start

    for (int kb = 0; kb < kbMax; ++kb) {
        __syncthreads();   // previous iteration's LDS readers done
        // ---- cooperative K/V tile load, f32 -> f16 ----
        {
            const float* kp = Kb + (size_t)(kb * KBLK + krow) * D_DIM + kcol;
            float4 a0 = *(const float4*)kp;
            float4 a1 = *(const float4*)(kp + 4);
            *(v8h*)(kT + krow * D_DIM + kcol) = pack8(a0, a1);

            const float* vp = Vb + (size_t)(kb * KBLK + krow) * D_DIM + kcol;
            float4 b0 = *(const float4*)vp;
            float4 b1 = *(const float4*)(vp + 4);
            float vv[8] = { b0.x, b0.y, b0.z, b0.w, b1.x, b1.y, b1.z, b1.w };
#pragma unroll
            for (int i = 0; i < 8; ++i)
                vT[(kcol + i) * VT_STRIDE + krow] = (_Float16)vv[i];

            if (kb + 1 < kbMax) {            // global_prefetch_b8 next tiles
                __builtin_prefetch(kp + KBLK * D_DIM, 0, 3);
                __builtin_prefetch(vp + KBLK * D_DIM, 0, 3);
            }
        }
        __syncthreads();

        // waves fully above the diagonal skip compute (still barrier)
        if (kb * KBLK <= qw + 15) {
            float p0[8], p1[8];
            // ---- S = Q K^T for two 16-key subtiles ----
#pragma unroll
            for (int j = 0; j < 2; ++j) {
                // B 32x16, k order chosen here to match A: lane ln = col (key); lanes<16 hold kk 0-15,
                // lanes>=16 hold kk 16-31 => 16 contiguous halfs of K's row.
                const _Float16* kr = kT + (j * 16 + ln) * D_DIM + hf * 16;
                v16h b0 = *(const v16h*)(kr);
                v16h b1 = *(const v16h*)(kr + 32);
                v8f  sc = {};
                sc = wmma_f16(qa[0], b0, sc);
                sc = wmma_f16(qa[1], b1, sc);
                float* pj = j ? p1 : p0;
#pragma unroll
                for (int r = 0; r < 8; ++r) pj[r] = sc[r];
            }

            // ---- causal mask: only the wave's diagonal block needs it ----
            if (kb * KBLK + (KBLK - 1) > qw) {       // wave-uniform branch
                const int qr0 = qw + hf * 8;
#pragma unroll
                for (int r = 0; r < 8; ++r) {
                    const int q = qr0 + r;
                    if (kb * KBLK + ln > q)      p0[r] = -1e30f;
                    if (kb * KBLK + 16 + ln > q) p1[r] = -1e30f;
                }
            }

            // ---- online softmax: running max + rescale ----
            float scale[8];
#pragma unroll
            for (int r = 0; r < 8; ++r) {
                float mx = rmax16(fmaxf(p0[r], p1[r]));
                float mn = fmaxf(m[r], mx);
                scale[r] = __expf(m[r] - mn);
                m[r] = mn;
                p0[r] = __expf(p0[r] - mn);
                p1[r] = __expf(p1[r] - mn);
            }
#pragma unroll
            for (int t = 0; t < 4; ++t)
#pragma unroll
                for (int r = 0; r < 8; ++r) o[t][r] *= scale[r];

            // ---- stash P (f16) and reload in A-fragment layout ----
            _Float16* pw = pS[w];
#pragma unroll
            for (int r = 0; r < 8; ++r) {
                pw[(r + hf * 8) * PS_STRIDE + ln]      = (_Float16)p0[r];
                pw[(r + hf * 8) * PS_STRIDE + 16 + ln] = (_Float16)p1[r];
            }
            asm volatile("s_wait_dscnt 0x0" ::: "memory");
            v8h plo = *(const v8h*)(pw + ln * PS_STRIDE + hf * 16);
            v8h phi = *(const v8h*)(pw + ln * PS_STRIDE + hf * 16 + 8);
            v16h pa = __builtin_shufflevector(plo, phi,
                        0, 1, 2, 3, 4, 5, 6, 7, 8, 9, 10, 11, 12, 13, 14, 15);

            // ---- l update: row sums of exp(P) via WMMA against ones ----
            v8f ls = {};
            ls = wmma_f16(pa, ones, ls);
#pragma unroll
            for (int r = 0; r < 8; ++r) l[r] = l[r] * scale[r] + ls[r];

            // ---- O += P V : B fragment from transposed V tile ----
#pragma unroll
            for (int t = 0; t < 4; ++t) {
                v16h vb = *(const v16h*)(vT + (t * 16 + ln) * VT_STRIDE + hf * 16);
                o[t] = wmma_f16(pa, vb, o[t]);
            }
        }
    }

    // ---- epilogue: normalize and store fp32 ----
#pragma unroll
    for (int r = 0; r < 8; ++r) {
        const float inv = 1.0f / l[r];
        const size_t qr = (size_t)(qw + r + hf * 8) * D_DIM;
#pragma unroll
        for (int t = 0; t < 4; ++t)
            Ob[qr + t * 16 + ln] = o[t][r] * inv;
    }
}

extern "C" void kernel_launch(void* const* d_in, const int* in_sizes, int n_in,
                              void* d_out, int out_size, void* d_ws, size_t ws_size,
                              hipStream_t stream) {
    (void)in_sizes; (void)n_in; (void)d_ws; (void)ws_size; (void)out_size;
    const float* Q = (const float*)d_in[0];
    const float* K = (const float*)d_in[1];
    const float* V = (const float*)d_in[2];
    // d_in[3] is the causal mask; causality is applied analytically.
    float* out = (float*)d_out;

    dim3 grid(S_LEN / BLOCK_M, NBH);   // 16 x 64
    dim3 block(256);                   // 8 wave32 waves
    fa_fwd_kernel<<<grid, block, 0, stream>>>(Q, K, V, out);
}
